// FNN_Back3_49486613184978
// MI455X (gfx1250) — matrix operand order repaired and checked by reading, not compiled, not run on hardware
//
#include <hip/hip_runtime.h>
#include <hip/hip_bf16.h>

// ---------------------------------------------------------------------------
// Foveated CNN forward, MI455X (gfx1250, wave32, WMMA).
// NHWC fp16 activations (channels padded to >=8), weights repacked to
// (cout, tap, cin_pad) with K zero-padded to a multiple of 32 so every WMMA
// fragment 8-run is one contiguous 16-byte load.  Each wave computes a
// 16-pixel column for ALL output channels (CT accumulators), amortizing the
// im2col gather.  fp32 accumulate, bias fused, fused stitch+ReLU+maxpool.
// ---------------------------------------------------------------------------

typedef __attribute__((ext_vector_type(16))) _Float16 v16h;
typedef __attribute__((ext_vector_type(8)))  _Float16 v8h;
typedef __attribute__((ext_vector_type(8)))  float    v8f;

// --------- input conversion: (B,3,H,W) f32 -> (B,H,W,8) f16 (pad ch) -------
__global__ void cvt_x_nhwc(const float* __restrict__ x,
                           _Float16* __restrict__ d, int B_, int H, int W) {
    int idx = blockIdx.x * blockDim.x + threadIdx.x;
    int total = B_ * H * W * 8;
    if (idx >= total) return;
    int c = idx & 7;
    int pix = idx >> 3;
    int xx = pix % W;
    int t = pix / W;
    int y = t % H;
    int b = t / H;
    float v = (c < 3) ? x[(((size_t)b * 3 + c) * H + y) * W + xx] : 0.f;
    d[idx] = (_Float16)v;
}

// --------- weight repack: (OC,CinReal,KS,KS) f32 -> (OC, KPAD) f16 ---------
// K order = tap-major, cin innermost (tap = kh*KS+kw), zero padded.
__global__ void repack_weights(const float* __restrict__ src,
                               _Float16* __restrict__ dst, int OC, int CinReal,
                               int CIN, int TAPS, int KPAD) {
    int idx = blockIdx.x * blockDim.x + threadIdx.x;
    int total = OC * KPAD;
    if (idx >= total) return;
    int K = idx % KPAD;
    int oc = idx / KPAD;
    int tap = K / CIN;
    int c = K - tap * CIN;
    float v = 0.f;
    if (tap < TAPS && c < CinReal)
        v = src[((size_t)oc * CinReal + c) * TAPS + tap];
    dst[idx] = (_Float16)v;
}

// ----------------------- bilinear resize, NHWC f16 --------------------------
__global__ void resize_bilinear_nhwc(const _Float16* __restrict__ src,
                                     _Float16* __restrict__ dst, int B_, int C,
                                     int Hs, int Ws, int Hd, int Wd) {
    int idx = blockIdx.x * blockDim.x + threadIdx.x;
    int total = B_ * Hd * Wd * C;
    if (idx >= total) return;
    int c = idx % C;
    int t = idx / C;
    int x = t % Wd;
    t /= Wd;
    int y = t % Hd;
    int b = t / Hd;
    float sy = ((y + 0.5f) * (float)Hs) / (float)Hd - 0.5f;
    float sx = ((x + 0.5f) * (float)Ws) / (float)Wd - 0.5f;
    sy = fminf(fmaxf(sy, 0.f), (float)(Hs - 1));
    sx = fminf(fmaxf(sx, 0.f), (float)(Ws - 1));
    int y0 = (int)sy, x0 = (int)sx;
    int y1 = min(y0 + 1, Hs - 1), x1 = min(x0 + 1, Ws - 1);
    float fy = sy - (float)y0, fx = sx - (float)x0;
    const _Float16* p = src + (size_t)b * Hs * Ws * C + c;
    float v00 = (float)p[((size_t)y0 * Ws + x0) * C];
    float v01 = (float)p[((size_t)y0 * Ws + x1) * C];
    float v10 = (float)p[((size_t)y1 * Ws + x0) * C];
    float v11 = (float)p[((size_t)y1 * Ws + x1) * C];
    float v = (v00 * (1.f - fy) + v10 * fy) * (1.f - fx) +
              (v01 * (1.f - fy) + v11 * fy) * fx;
    dst[idx] = (_Float16)v;
}

// ---------------- WMMA implicit-GEMM conv ('same'), NHWC --------------------
// One wave -> 16 pixels x COUT channels.  K chunks of 32 over (tap, cin).
template <int KS, int CIN, int COUT>
__global__ __launch_bounds__(128) void conv_wmma_nhwc(
    const _Float16* __restrict__ in,   // (B,H,W,CIN)
    const _Float16* __restrict__ wgt,  // (COUT, KPAD) f16
    const float* __restrict__ bias,    // (COUT)
    _Float16* __restrict__ out,        // (B,H,W,COUT), pre-ReLU, bias added
    int H, int W) {
    constexpr int PAD = KS / 2;
    constexpr int TAPS = KS * KS;
    constexpr int NCH = (TAPS * CIN + 31) / 32;  // K chunks of 32
    constexpr int KPAD = NCH * 32;
    constexpr int CT = COUT / 16;

    const int lane = threadIdx.x & 31;
    const int wv = threadIdx.x >> 5;
    const int hi = lane >> 4;
    const int n = lane & 15;
    const int b = blockIdx.z;
    const int HW = H * W;
    const int p = (blockIdx.x * 4 + wv) * 16 + n;
    const bool valid = (p < HW);
    const int oy = valid ? (p / W) : 0;
    const int ox = valid ? (p % W) : 0;

    const _Float16* inB = in + (size_t)b * HW * CIN;
    __builtin_prefetch(wgt + (size_t)n * KPAD, 0, 1);  // global_prefetch_b8

    v8f acc[CT] = {};

    for (int q = 0; q < NCH; ++q) {
        const int k0 = q * 32;
        // --- B fragment: lane n = pixel column; two contiguous 8-runs ---
        v8h bv0 = {}, bv1 = {};
#pragma unroll
        for (int j = 0; j < 2; ++j) {
            int K8 = k0 + hi * 16 + j * 8;
            int tap = K8 / CIN;  // CIN is power of two -> shift
            int c0 = K8 - tap * CIN;
            int kh = tap / KS;
            int kw = tap - kh * KS;
            int iy = oy + kh - PAD;
            int ix = ox + kw - PAD;
            bool ok = valid && (tap < TAPS) && (unsigned)iy < (unsigned)H &&
                      (unsigned)ix < (unsigned)W;
            v8h v = {};
            if (ok) v = *(const v8h*)(inB + ((size_t)iy * W + ix) * CIN + c0);
            if (j == 0) bv0 = v; else bv1 = v;
        }
        v16h bfrag = __builtin_shufflevector(bv0, bv1, 0, 1, 2, 3, 4, 5, 6, 7,
                                             8, 9, 10, 11, 12, 13, 14, 15);
        // --- A fragments (weights), one WMMA per cout tile ---
#pragma unroll
        for (int ct = 0; ct < CT; ++ct) {
            const _Float16* wr =
                wgt + (size_t)(ct * 16 + n) * KPAD + k0 + hi * 16;
            v8h a0 = *(const v8h*)wr;
            v8h a1 = *(const v8h*)(wr + 8);
            v16h afrag = __builtin_shufflevector(a0, a1, 0, 1, 2, 3, 4, 5, 6,
                                                 7, 8, 9, 10, 11, 12, 13, 14,
                                                 15);
            acc[ct] = __builtin_amdgcn_wmma_f32_16x16x32_f16(
                false, afrag, false, bfrag, (short)0, acc[ct], false, false);
        }
    }

    if (valid) {
        _Float16* op = out + (((size_t)b * H + oy) * W + ox) * COUT;
#pragma unroll
        for (int ct = 0; ct < CT; ++ct) {
            v8h r;
#pragma unroll
            for (int v = 0; v < 8; ++v)
                r[v] = (_Float16)(acc[ct][v] + bias[ct * 16 + hi * 8 + v]);
            *(v8h*)(op + ct * 16 + hi * 8) = r;  // 8 consecutive couts
        }
    }
}

// ------------------- on-the-fly bilinear upsample, NHWC ---------------------
__device__ inline float upsample_nhwc(const _Float16* __restrict__ base,
                                      int Hs, int Ws, int C, int c, int Hd,
                                      int Wd, int y, int x) {
    float sy = ((y + 0.5f) * (float)Hs) / (float)Hd - 0.5f;
    float sx = ((x + 0.5f) * (float)Ws) / (float)Wd - 0.5f;
    sy = fminf(fmaxf(sy, 0.f), (float)(Hs - 1));
    sx = fminf(fmaxf(sx, 0.f), (float)(Ws - 1));
    int y0 = (int)sy, x0 = (int)sx;
    int y1 = min(y0 + 1, Hs - 1), x1 = min(x0 + 1, Ws - 1);
    float fy = sy - (float)y0, fx = sx - (float)x0;
    float v00 = (float)base[((size_t)y0 * Ws + x0) * C + c];
    float v01 = (float)base[((size_t)y0 * Ws + x1) * C + c];
    float v10 = (float)base[((size_t)y1 * Ws + x0) * C + c];
    float v11 = (float)base[((size_t)y1 * Ws + x1) * C + c];
    return (v00 * (1.f - fy) + v10 * fy) * (1.f - fx) +
           (v01 * (1.f - fy) + v11 * fy) * fx;
}

// -------------- fused stitch + ReLU + 2x2 maxpool, NHWC f16 -----------------
__global__ void stitch_pool_nhwc(const _Float16* __restrict__ y0,
                                 const _Float16* __restrict__ y1,
                                 const _Float16* __restrict__ y2,
                                 const int* __restrict__ foa,
                                 _Float16* __restrict__ out, int B_, int C,
                                 int H, int W, int H1, int W1, int H2, int W2,
                                 float t0, float t1, float foaScale) {
    int Ho = H >> 1, Wo = W >> 1;
    int idx = blockIdx.x * blockDim.x + threadIdx.x;
    int total = B_ * Ho * Wo * C;
    if (idx >= total) return;
    int c = idx % C;
    int t = idx / C;
    int px = t % Wo;
    t /= Wo;
    int py = t % Ho;
    int b = t / Ho;

    float fr = floorf((float)foa[2 * b] * foaScale);
    float fc = floorf((float)foa[2 * b + 1] * foaScale);

    const _Float16* p0 = y0 + (size_t)b * H * W * C;
    const _Float16* p1 = y1 + (size_t)b * H1 * W1 * C;
    const _Float16* p2 = y2 + (size_t)b * H2 * W2 * C;

    float mx = 0.f;  // max(0, max v) == max of ReLU'd values
#pragma unroll
    for (int dy = 0; dy < 2; ++dy) {
#pragma unroll
        for (int dx = 0; dx < 2; ++dx) {
            int oy = 2 * py + dy, ox = 2 * px + dx;
            float dist = fmaxf(fabsf((float)oy - fr), fabsf((float)ox - fc));
            float v;
            if (dist <= t0)      v = (float)p0[((size_t)oy * W + ox) * C + c];
            else if (dist <= t1) v = upsample_nhwc(p1, H1, W1, C, c, H, W, oy, ox);
            else                 v = upsample_nhwc(p2, H2, W2, C, c, H, W, oy, ox);
            mx = fmaxf(mx, v);
        }
    }
    out[idx] = (_Float16)mx;
}

// ------- final: stitch + ReLU -> NCHW fp32 d_out, region index int32 --------
__global__ void stitch_final_nhwc(const _Float16* __restrict__ y0,
                                  const _Float16* __restrict__ y1,
                                  const int* __restrict__ foa,
                                  float* __restrict__ out,
                                  int* __restrict__ ind, int B_, int C, int H,
                                  int W, int H1, int W1, float t0,
                                  float foaScale) {
    int idx = blockIdx.x * blockDim.x + threadIdx.x;
    int total = B_ * H * W;
    if (idx >= total) return;
    int ox = idx % W;
    int t = idx / W;
    int oy = t % H;
    int b = t / H;

    float fr = floorf((float)foa[2 * b] * foaScale);
    float fc = floorf((float)foa[2 * b + 1] * foaScale);
    float dist = fmaxf(fabsf((float)oy - fr), fabsf((float)ox - fc));
    int k = (dist <= t0) ? 0 : 1;
    ind[idx] = k;

    const _Float16* p0 = y0 + (((size_t)b * H + oy) * W + ox) * C;
    const _Float16* p1 = y1 + (size_t)b * H1 * W1 * C;

    // hoist bilinear coords for the low-res bank
    float sy = ((oy + 0.5f) * (float)H1) / (float)H - 0.5f;
    float sx = ((ox + 0.5f) * (float)W1) / (float)W - 0.5f;
    sy = fminf(fmaxf(sy, 0.f), (float)(H1 - 1));
    sx = fminf(fmaxf(sx, 0.f), (float)(W1 - 1));
    int y0i = (int)sy, x0i = (int)sx;
    int y1i = min(y0i + 1, H1 - 1), x1i = min(x0i + 1, W1 - 1);
    float fy = sy - (float)y0i, fx = sx - (float)x0i;
    const _Float16* q00 = p1 + ((size_t)y0i * W1 + x0i) * C;
    const _Float16* q01 = p1 + ((size_t)y0i * W1 + x1i) * C;
    const _Float16* q10 = p1 + ((size_t)y1i * W1 + x0i) * C;
    const _Float16* q11 = p1 + ((size_t)y1i * W1 + x1i) * C;

    for (int c = 0; c < C; ++c) {
        float v;
        if (k == 0) {
            v = (float)p0[c];
        } else {
            v = ((float)q00[c] * (1.f - fy) + (float)q10[c] * fy) * (1.f - fx) +
                ((float)q01[c] * (1.f - fy) + (float)q11[c] * fy) * fx;
        }
        out[(((size_t)b * C + c) * H + oy) * W + ox] = fmaxf(v, 0.f);
    }
}

// ---------------------------------------------------------------------------
extern "C" void kernel_launch(void* const* d_in, const int* in_sizes, int n_in,
                              void* d_out, int out_size, void* d_ws,
                              size_t ws_size, hipStream_t stream) {
    (void)in_sizes; (void)n_in; (void)out_size; (void)ws_size;
    const float* x = (const float*)d_in[0];
    const int* foa = (const int*)d_in[1];
    const float* w1 = (const float*)d_in[2];
    const float* b1 = (const float*)d_in[3];
    const float* w2 = (const float*)d_in[4];
    const float* b2 = (const float*)d_in[5];
    const float* w3 = (const float*)d_in[6];
    const float* b3 = (const float*)d_in[7];
    const float* w4 = (const float*)d_in[8];
    const float* b4 = (const float*)d_in[9];

    const int B = 16;
    // padded K sizes: ((KS*KS*CIN + 31)/32)*32
    const int KP1 = 224;  // 5x5, CIN=8   (200 -> 224)
    const int KP2 = 416;  // 5x5, CIN=16  (400 -> 416)
    const int KP3 = 288;  // 3x3, CIN=32
    const int KP4 = 576;  // 3x3, CIN=64

    char* ws = (char*)d_ws;
    size_t off = 0;
    auto alloc = [&](size_t elems) -> _Float16* {
        _Float16* p = (_Float16*)(ws + off);
        off += ((elems * sizeof(_Float16)) + 255) & ~(size_t)255;
        return p;
    };

    // NHWC fp16 activations / bank outputs
    _Float16* a0_1 = alloc((size_t)B * 224 * 224 * 8);
    _Float16* a1_1 = alloc((size_t)B * 112 * 112 * 8);
    _Float16* a2_1 = alloc((size_t)B * 56 * 56 * 8);
    _Float16* y0_1 = alloc((size_t)B * 224 * 224 * 16);
    _Float16* y1_1 = alloc((size_t)B * 112 * 112 * 16);
    _Float16* y2_1 = alloc((size_t)B * 56 * 56 * 16);
    _Float16* p1   = alloc((size_t)B * 112 * 112 * 16);
    _Float16* a1_2 = alloc((size_t)B * 56 * 56 * 16);
    _Float16* a2_2 = alloc((size_t)B * 28 * 28 * 16);
    _Float16* y0_2 = alloc((size_t)B * 112 * 112 * 32);
    _Float16* y1_2 = alloc((size_t)B * 56 * 56 * 32);
    _Float16* y2_2 = alloc((size_t)B * 28 * 28 * 32);
    _Float16* p2   = alloc((size_t)B * 56 * 56 * 32);
    _Float16* a1_3 = alloc((size_t)B * 14 * 14 * 32);
    _Float16* y0_3 = alloc((size_t)B * 56 * 56 * 64);
    _Float16* y1_3 = alloc((size_t)B * 14 * 14 * 64);
    _Float16* p3   = alloc((size_t)B * 28 * 28 * 64);
    _Float16* a1_4 = alloc((size_t)B * 7 * 7 * 64);
    _Float16* y0_4 = alloc((size_t)B * 28 * 28 * 128);
    _Float16* y1_4 = alloc((size_t)B * 7 * 7 * 128);
    // repacked fp16 weights
    _Float16* w1h = alloc((size_t)3 * 16 * KP1);
    _Float16* w2h = alloc((size_t)3 * 32 * KP2);
    _Float16* w3h = alloc((size_t)2 * 64 * KP3);
    _Float16* w4h = alloc((size_t)2 * 128 * KP4);

    auto repack = [&](const float* s, _Float16* d, int OC, int CinReal,
                      int CIN, int TAPS, int KPAD) {
        int n = OC * KPAD;
        repack_weights<<<(n + 255) / 256, 256, 0, stream>>>(s, d, OC, CinReal,
                                                            CIN, TAPS, KPAD);
    };
    auto resize = [&](const _Float16* s, _Float16* d, int C, int Hs, int Hd) {
        int n = B * Hd * Hd * C;
        resize_bilinear_nhwc<<<(n + 255) / 256, 256, 0, stream>>>(
            s, d, B, C, Hs, Hs, Hd, Hd);
    };

    // ---- weight & input conversion ----
    {
        int n = B * 224 * 224 * 8;
        cvt_x_nhwc<<<(n + 255) / 256, 256, 0, stream>>>(x, a0_1, B, 224, 224);
    }
    repack(w1, w1h, 3 * 16, 3, 8, 25, KP1);
    repack(w2, w2h, 3 * 32, 16, 16, 25, KP2);
    repack(w3, w3h, 2 * 64, 32, 32, 9, KP3);
    repack(w4, w4h, 2 * 128, 64, 64, 9, KP4);

    auto cgrid = [&](int H) { return dim3((H * H + 63) / 64, 1, B); };

    // ---- layer 1: 3 banks, 5x5, 8(pad3) -> 16 @224 ----
    resize(a0_1, a1_1, 8, 224, 112);
    resize(a0_1, a2_1, 8, 224, 56);
    conv_wmma_nhwc<5, 8, 16><<<cgrid(224), 128, 0, stream>>>(
        a0_1, w1h + 0 * 16 * KP1, b1, y0_1, 224, 224);
    conv_wmma_nhwc<5, 8, 16><<<cgrid(112), 128, 0, stream>>>(
        a1_1, w1h + 1 * 16 * KP1, b1, y1_1, 112, 112);
    conv_wmma_nhwc<5, 8, 16><<<cgrid(56), 128, 0, stream>>>(
        a2_1, w1h + 2 * 16 * KP1, b1, y2_1, 56, 56);
    {
        int n = B * 112 * 112 * 16;
        stitch_pool_nhwc<<<(n + 255) / 256, 256, 0, stream>>>(
            y0_1, y1_1, y2_1, foa, p1, B, 16, 224, 224, 112, 112, 56, 56,
            25.f, 50.f, 1.0f);
    }

    // ---- layer 2: 3 banks, 5x5, 16 -> 32 @112 ----
    resize(p1, a1_2, 16, 112, 56);
    resize(p1, a2_2, 16, 112, 28);
    conv_wmma_nhwc<5, 16, 32><<<cgrid(112), 128, 0, stream>>>(
        p1, w2h + 0 * 32 * KP2, b2, y0_2, 112, 112);
    conv_wmma_nhwc<5, 16, 32><<<cgrid(56), 128, 0, stream>>>(
        a1_2, w2h + 1 * 32 * KP2, b2, y1_2, 56, 56);
    conv_wmma_nhwc<5, 16, 32><<<cgrid(28), 128, 0, stream>>>(
        a2_2, w2h + 2 * 32 * KP2, b2, y2_2, 28, 28);
    {
        int n = B * 56 * 56 * 32;
        stitch_pool_nhwc<<<(n + 255) / 256, 256, 0, stream>>>(
            y0_2, y1_2, y2_2, foa, p2, B, 32, 112, 112, 56, 56, 28, 28, 12.f,
            25.f, 0.5f);
    }

    // ---- layer 3: 2 banks, 3x3, 32 -> 64 @56 ----
    resize(p2, a1_3, 32, 56, 14);
    conv_wmma_nhwc<3, 32, 64><<<cgrid(56), 128, 0, stream>>>(
        p2, w3h + 0 * 64 * KP3, b3, y0_3, 56, 56);
    conv_wmma_nhwc<3, 32, 64><<<cgrid(14), 128, 0, stream>>>(
        a1_3, w3h + 1 * 64 * KP3, b3, y1_3, 14, 14);
    {
        int n = B * 28 * 28 * 64;
        stitch_pool_nhwc<<<(n + 255) / 256, 256, 0, stream>>>(
            y0_3, y1_3, y1_3, foa, p3, B, 64, 56, 56, 14, 14, 14, 14, 12.f,
            1e30f, 0.25f);
    }

    // ---- layer 4: 2 banks, 3x3, 64 -> 128 @28, no pool ----
    resize(p3, a1_4, 64, 28, 7);
    conv_wmma_nhwc<3, 64, 128><<<cgrid(28), 128, 0, stream>>>(
        p3, w4h + 0 * 128 * KP4, b4, y0_4, 28, 28);
    conv_wmma_nhwc<3, 64, 128><<<cgrid(7), 128, 0, stream>>>(
        a1_4, w4h + 1 * 128 * KP4, b4, y1_4, 7, 7);
    {
        float* out_f = (float*)d_out;
        int* ind = (int*)(out_f + (size_t)B * 128 * 28 * 28);
        int n = B * 28 * 28;
        stitch_final_nhwc<<<(n + 255) / 256, 256, 0, stream>>>(
            y0_4, y1_4, foa, out_f, ind, B, 128, 28, 28, 7, 7, 5.f, 0.125f);
    }
}
